// ACOPFPredictor_53747220742610
// MI455X (gfx1250) — matrix operand order repaired and checked by reading, not compiled, not run on hardware
//
#include <hip/hip_runtime.h>

// ---------------------------------------------------------------------------
// Problem constants (from the reference)
// ---------------------------------------------------------------------------
#define NBUS   100000
#define NGEN   20000
#define NLOAD  60000
#define NSHUNT 5000
#define DIN    32
#define HID    256
#define NOUT   4
#define NE_AC  150000
#define NE_TR  30000

typedef __attribute__((ext_vector_type(16))) __bf16 v16bf;
typedef __attribute__((ext_vector_type(8)))  __bf16 v8bf;
typedef __attribute__((ext_vector_type(8)))  float  v8f;

__device__ __forceinline__ __bf16 f2bf(float f) {
  union { unsigned int u; float f; } in; in.f = f;
  unsigned int x = in.u;
  unsigned int r = x + 0x7FFFu + ((x >> 16) & 1u);   // round-to-nearest-even
  union { unsigned short s; __bf16 b; } out;
  out.s = (unsigned short)(r >> 16);
  return out.b;
}

// ---------------------------------------------------------------------------
// Elementwise utility kernels
// ---------------------------------------------------------------------------
__global__ void fill_f32(float* __restrict__ p, float v, int n) {
  int i = blockIdx.x * blockDim.x + threadIdx.x;
  if (i < n) p[i] = v;
}

__global__ void f32_to_bf16_k(const float* __restrict__ x, __bf16* __restrict__ y, int n) {
  int i = blockIdx.x * blockDim.x + threadIdx.x;
  if (i < n) y[i] = f2bf(x[i]);
}

__global__ void relu_k(float* __restrict__ x, int n) {
  int i = blockIdx.x * blockDim.x + threadIdx.x;
  if (i < n) x[i] = fmaxf(x[i], 0.0f);
}

__global__ void rsqrt_k(float* __restrict__ p, int n) {
  int i = blockIdx.x * blockDim.x + threadIdx.x;
  if (i < n) p[i] = (p[i] > 0.0f) ? rsqrtf(p[i]) : 0.0f;
}

__global__ void scatter_count(const int* __restrict__ dst, int E, float* __restrict__ cnt) {
  int e = blockIdx.x * blockDim.x + threadIdx.x;
  if (e < E) atomicAdd(&cnt[dst[e]], 1.0f);
}

// ---------------------------------------------------------------------------
// Weight packing: fp32 W[K,HID] row-major -> bf16 B-fragment layout.
// Fragment tile t = kt*(HID/16)+nt ; within tile, lane-major, 16 elems/lane.
// Element e of lane L maps to K = kt*32 + (L>>4)*8 + e (e<8) or kt*32 + 16 + (L>>4)*8 + (e-8) (e>=8), N = nt*16 + (L&15)
// (ISA 7.12.2 16-bit B-matrix layout for 16x16x32).
// ---------------------------------------------------------------------------
__global__ void pack_B(const float* __restrict__ W, __bf16* __restrict__ Bp, int K) {
  int idx = blockIdx.x * blockDim.x + threadIdx.x;
  if (idx >= K * HID) return;
  int e    = idx & 15;
  int lane = (idx >> 4) & 31;
  int t    = idx >> 9;
  const int NT = HID / 16;
  int nt = t % NT;
  int kt = t / NT;
  int kk = kt * 32 + (lane >> 4) * 8 + (e < 8 ? e : 16 + (e - 8));
  int nn = nt * 16 + (lane & 15);
  Bp[idx] = f2bf(W[kk * HID + nn]);
}

// ---------------------------------------------------------------------------
// WMMA GEMM:  C[M,HID] (+)= A_bf16[M,K] @ W_packed  (+ bias)
// Block = 256 threads = 8 waves. Wave computes a 16x64 tile of C:
//   4 accumulators, 4 x v_wmma_f32_16x16x32_bf16 per K-step.
// A fragment per lane: rows M=mtile*16+(lane&15);
//   lanes 0-15 hold K = {0..7, 16..23}, lanes 16-31 hold K = {8..15, 24..31}
//   => two contiguous 16B bf16 loads.
// B fragment per lane: one contiguous 32B load from the packed buffer.
// ---------------------------------------------------------------------------
__global__ __launch_bounds__(256) void gemm_bf16_wmma(
    const __bf16* __restrict__ A, const __bf16* __restrict__ Bp,
    float* __restrict__ C, const float* __restrict__ bias,
    int M, int K, int accumulate)
{
  const int wave = threadIdx.x >> 5;
  const int lane = threadIdx.x & 31;
  const int mtile = blockIdx.x * 2 + (wave >> 2);
  if (mtile * 16 >= M) return;                // wave-uniform: EXEC stays full
  const int nq = wave & 3;                    // 64-column group
  const int NT = HID / 16;

  int rowA = mtile * 16 + (lane & 15);
  if (rowA >= M) rowA = M - 1;                // clamp: loads valid, store masked
  const int kb = (lane >> 4) * 8;
  const __bf16* aptr = A + (long)rowA * K + kb;

  v8f acc[4];
#pragma unroll
  for (int j = 0; j < 4; ++j)
#pragma unroll
    for (int r = 0; r < 8; ++r) acc[j][r] = 0.0f;

  const int KT = K >> 5;
  for (int kt = 0; kt < KT; ++kt) {
    const v8bf* a0 = (const v8bf*)(aptr + kt * 32);
    v8bf lo = a0[0];
    v8bf hi = a0[2];                          // +16 elements
    v16bf afrag;
#pragma unroll
    for (int i = 0; i < 8; ++i) { afrag[i] = lo[i]; afrag[8 + i] = hi[i]; }
#pragma unroll
    for (int j = 0; j < 4; ++j) {
      int nt = nq * 4 + j;
      v16bf bfrag = *(const v16bf*)(Bp + ((long)(kt * NT + nt) * 32 + lane) * 16);
      acc[j] = __builtin_amdgcn_wmma_f32_16x16x32_bf16(
          false, afrag, false, bfrag, (short)0, acc[j], false, false);
    }
  }

  // C/D layout: VGPR r -> lanes 0-15: M=r, lanes 16-31: M=8+r ; N=lane&15
  const int mrow0 = mtile * 16 + (lane >> 4) * 8;
  const int ncol  = lane & 15;
#pragma unroll
  for (int j = 0; j < 4; ++j) {
    int n = (nq * 4 + j) * 16 + ncol;
    float bv = bias ? bias[n] : 0.0f;
#pragma unroll
    for (int r = 0; r < 8; ++r) {
      int m = mrow0 + r;
      if (m < M) {
        long off = (long)m * HID + n;
        float v = acc[j][r] + bv;
        if (accumulate) C[off] += v; else C[off] = v;
      }
    }
  }
}

// ---------------------------------------------------------------------------
// GCN edge scatter:  acc[dst] += h[src] * dinv[src]*dinv[dst]   (HID cols)
// ---------------------------------------------------------------------------
__global__ void gcn_scatter(const float* __restrict__ h, const int* __restrict__ src,
                            const int* __restrict__ dst, const float* __restrict__ dinv,
                            float* __restrict__ acc, int E)
{
  int tid = blockIdx.x * blockDim.x + threadIdx.x;
  if (tid >= E * 64) return;
  int e = tid >> 6;
  int c = (tid & 63) << 2;
  int s = src[e], d = dst[e];
  float nrm = dinv[s] * dinv[d];
  const float* hp = h + (long)s * HID + c;
  float* ap = acc + (long)d * HID + c;
  atomicAdd(ap + 0, hp[0] * nrm);
  atomicAdd(ap + 1, hp[1] * nrm);
  atomicAdd(ap + 2, hp[2] * nrm);
  atomicAdd(ap + 3, hp[3] * nrm);
}

// GCN self loop + bias:  acc[i,c] += h[i,c]*dinv[i]^2 + bias[c]
__global__ void gcn_self_bias(const float* __restrict__ h, const float* __restrict__ dinv,
                              const float* __restrict__ bias, float* __restrict__ acc, int N)
{
  int idx = blockIdx.x * blockDim.x + threadIdx.x;
  if (idx >= N * HID) return;
  int i = idx >> 8;
  int c = idx & (HID - 1);
  float dv = dinv[i];
  acc[idx] += h[idx] * dv * dv + bias[c];
}

// SAGE neighbor sum:  agg[dst] += xsrc[src]   (d cols)
__global__ void sage_scatter(const float* __restrict__ xsrc, const int* __restrict__ src,
                             const int* __restrict__ dst, float* __restrict__ agg,
                             int E, int d)
{
  int per = d >> 2;
  int tid = blockIdx.x * blockDim.x + threadIdx.x;
  if (tid >= E * per) return;
  int e = tid / per;
  int c = (tid % per) << 2;
  int s = src[e], dd = dst[e];
  const float* xp = xsrc + (long)s * d + c;
  float* ap = agg + (long)dd * d + c;
  atomicAdd(ap + 0, xp[0]);
  atomicAdd(ap + 1, xp[1]);
  atomicAdd(ap + 2, xp[2]);
  atomicAdd(ap + 3, xp[3]);
}

// mean = agg / max(cnt,1), emitted as bf16 (GEMM A operand)
__global__ void mean_to_bf16(const float* __restrict__ agg, const float* __restrict__ cnt,
                             __bf16* __restrict__ y, int N, int d)
{
  int idx = blockIdx.x * blockDim.x + threadIdx.x;
  if (idx >= N * d) return;
  int i = idx / d;
  y[idx] = f2bf(agg[idx] / fmaxf(cnt[i], 1.0f));
}

// Final 256 -> 4 linear (tiny; VALU is fine here)
__global__ void final_lin(const float* __restrict__ x, const float* __restrict__ W,
                          const float* __restrict__ b, float* __restrict__ out, int N)
{
  int idx = blockIdx.x * blockDim.x + threadIdx.x;
  if (idx >= N * NOUT) return;
  int i = idx >> 2, o = idx & 3;
  float s = b[o];
  const float* xr = x + (long)i * HID;
#pragma unroll 8
  for (int k = 0; k < HID; ++k) s += xr[k] * W[k * NOUT + o];
  out[idx] = s;
}

// ---------------------------------------------------------------------------
// Host orchestration
// ---------------------------------------------------------------------------
struct SageP { const float *Wl, *Wr, *b; };
struct GcnP  { const float *W,  *b; };
struct LayerP { SageP b2g, b2l, b2s, g2b; GcnP gcn_ac, gcn_tr; SageP l2b, s2b; };

static inline int cdiv(int a, int b) { return (a + b - 1) / b; }

static void launch_gemm(const __bf16* A, const __bf16* Bp, float* C, const float* bias,
                        int M, int K, int accum, hipStream_t s) {
  gemm_bf16_wmma<<<cdiv(M, 32), 256, 0, s>>>(A, Bp, C, bias, M, K, accum);
}

static void run_gcn(const __bf16* xb_bus, const int* src, const int* dst, int E,
                    const GcnP& p, float* acc, float* hbuf, float* degbuf,
                    __bf16* wpack, int d, hipStream_t s)
{
  pack_B<<<cdiv(d * HID, 256), 256, 0, s>>>(p.W, wpack, d);
  launch_gemm(xb_bus, wpack, hbuf, nullptr, NBUS, d, 0, s);        // h = x @ W
  fill_f32<<<cdiv(NBUS, 256), 256, 0, s>>>(degbuf, 1.0f, NBUS);    // self loop
  scatter_count<<<cdiv(E, 256), 256, 0, s>>>(dst, E, degbuf);
  rsqrt_k<<<cdiv(NBUS, 256), 256, 0, s>>>(degbuf, NBUS);
  gcn_scatter<<<cdiv(E * 64, 256), 256, 0, s>>>(hbuf, src, dst, degbuf, acc, E);
  gcn_self_bias<<<cdiv(NBUS * HID, 256), 256, 0, s>>>(hbuf, degbuf, p.b, acc, NBUS);
}

static void run_sage(const float* xsrc_f32, const __bf16* xb_dst, int Ndst,
                     const int* src, const int* dst, int E, int d,
                     const SageP& p, float* acc,
                     float* aggbuf, float* cntbuf, __bf16* meanb, __bf16* wpack,
                     hipStream_t s)
{
  fill_f32<<<cdiv(Ndst * d, 256), 256, 0, s>>>(aggbuf, 0.0f, Ndst * d);
  fill_f32<<<cdiv(Ndst, 256), 256, 0, s>>>(cntbuf, 0.0f, Ndst);
  sage_scatter<<<cdiv(E * (d >> 2), 256), 256, 0, s>>>(xsrc_f32, src, dst, aggbuf, E, d);
  scatter_count<<<cdiv(E, 256), 256, 0, s>>>(dst, E, cntbuf);
  mean_to_bf16<<<cdiv(Ndst * d, 256), 256, 0, s>>>(aggbuf, cntbuf, meanb, Ndst, d);
  pack_B<<<cdiv(d * HID, 256), 256, 0, s>>>(p.Wl, wpack, d);
  launch_gemm(meanb, wpack, acc, p.b, Ndst, d, 1, s);              // += mean@Wl + b
  pack_B<<<cdiv(d * HID, 256), 256, 0, s>>>(p.Wr, wpack, d);
  launch_gemm(xb_dst, wpack, acc, nullptr, Ndst, d, 1, s);         // += x_dst@Wr
}

extern "C" void kernel_launch(void* const* d_in, const int* in_sizes, int n_in,
                              void* d_out, int out_size, void* d_ws, size_t ws_size,
                              hipStream_t stream)
{
  (void)in_sizes; (void)n_in; (void)out_size; (void)ws_size;
  const int Ns[4] = {NBUS, NGEN, NLOAD, NSHUNT};

  // ---- inputs ----
  const float* x0[4] = {(const float*)d_in[0], (const float*)d_in[1],
                        (const float*)d_in[2], (const float*)d_in[3]};
  const int* ac_s  = (const int*)d_in[4];  const int* ac_d  = (const int*)d_in[5];
  const int* tr_s  = (const int*)d_in[6];  const int* tr_d  = (const int*)d_in[7];
  const int* g2b_s = (const int*)d_in[8];  const int* g2b_d = (const int*)d_in[9];
  const int* b2g_s = (const int*)d_in[10]; const int* b2g_d = (const int*)d_in[11];
  const int* l2b_s = (const int*)d_in[12]; const int* l2b_d = (const int*)d_in[13];
  const int* b2l_s = (const int*)d_in[14]; const int* b2l_d = (const int*)d_in[15];
  const int* s2b_s = (const int*)d_in[16]; const int* s2b_d = (const int*)d_in[17];
  const int* b2s_s = (const int*)d_in[18]; const int* b2s_d = (const int*)d_in[19];

  // params: JAX pytree leaves, dicts in sorted-key order
  int pi = 20;
  auto nextp = [&]() { return (const float*)d_in[pi++]; };
  auto readSage = [&](SageP& sp) { sp.Wl = nextp(); sp.Wr = nextp(); sp.b = nextp(); };
  auto readGcn  = [&](GcnP& g)  { g.W = nextp(); g.b = nextp(); };
  auto readLayer = [&](LayerP& L) {
    readSage(L.b2g); readSage(L.b2l); readSage(L.b2s); readSage(L.g2b);
    readGcn(L.gcn_ac); readGcn(L.gcn_tr); readSage(L.l2b); readSage(L.s2b);
  };
  LayerP L1, L2;
  readLayer(L1); readLayer(L2);
  const float* linW = nextp();
  const float* linb = nextp();

  // ---- workspace carve-out (every buffer written before read each call) ----
  char* base = (char*)d_ws;
  size_t off = 0;
  auto alloc = [&](size_t bytes) -> void* {
    void* p = base + off; off = (off + bytes + 255) & ~(size_t)255; return p;
  };
  __bf16* xb[4];  for (int t = 0; t < 4; ++t) xb[t]  = (__bf16*)alloc((size_t)Ns[t] * HID * 2);
  float* actA[4]; for (int t = 0; t < 4; ++t) actA[t] = (float*)alloc((size_t)Ns[t] * HID * 4);
  float* actB[4]; for (int t = 0; t < 4; ++t) actB[t] = (float*)alloc((size_t)Ns[t] * HID * 4);
  float*  tmpf  = (float*) alloc((size_t)NBUS * HID * 4);   // shared: GCN h / SAGE agg
  __bf16* meanb = (__bf16*)alloc((size_t)NBUS * HID * 2);
  float*  cnt   = (float*) alloc((size_t)NBUS * 4);
  __bf16* wpack = (__bf16*)alloc((size_t)HID * HID * 2);

  // ---- one hetero layer ----
  auto run_layer = [&](const float* x[4], int d, const LayerP& L, float* acc[4]) {
    for (int t = 0; t < 4; ++t)
      f32_to_bf16_k<<<cdiv(Ns[t] * d, 256), 256, 0, stream>>>(x[t], xb[t], Ns[t] * d);
    for (int t = 0; t < 4; ++t)
      fill_f32<<<cdiv(Ns[t] * HID, 256), 256, 0, stream>>>(acc[t], 0.0f, Ns[t] * HID);

    run_gcn(xb[0], ac_s, ac_d, NE_AC, L.gcn_ac, acc[0], tmpf, cnt, wpack, d, stream);
    run_gcn(xb[0], tr_s, tr_d, NE_TR, L.gcn_tr, acc[0], tmpf, cnt, wpack, d, stream);

    run_sage(x[1], xb[0], NBUS,  g2b_s, g2b_d, NGEN,   d, L.g2b, acc[0], tmpf, cnt, meanb, wpack, stream);
    run_sage(x[2], xb[0], NBUS,  l2b_s, l2b_d, NLOAD,  d, L.l2b, acc[0], tmpf, cnt, meanb, wpack, stream);
    run_sage(x[3], xb[0], NBUS,  s2b_s, s2b_d, NSHUNT, d, L.s2b, acc[0], tmpf, cnt, meanb, wpack, stream);
    run_sage(x[0], xb[1], NGEN,  b2g_s, b2g_d, NGEN,   d, L.b2g, acc[1], tmpf, cnt, meanb, wpack, stream);
    run_sage(x[0], xb[2], NLOAD, b2l_s, b2l_d, NLOAD,  d, L.b2l, acc[2], tmpf, cnt, meanb, wpack, stream);
    run_sage(x[0], xb[3], NSHUNT,b2s_s, b2s_d, NSHUNT, d, L.b2s, acc[3], tmpf, cnt, meanb, wpack, stream);

    for (int t = 0; t < 4; ++t)
      relu_k<<<cdiv(Ns[t] * HID, 256), 256, 0, stream>>>(acc[t], Ns[t] * HID);
  };

  run_layer(x0, DIN, L1, actA);
  const float* x1[4] = {actA[0], actA[1], actA[2], actA[3]};
  run_layer(x1, HID, L2, actB);

  // ---- final linear, outputs concatenated (bus, gen, load, shunt) ----
  float* out = (float*)d_out;
  long ofs = 0;
  for (int t = 0; t < 4; ++t) {
    final_lin<<<cdiv(Ns[t] * NOUT, 256), 256, 0, stream>>>(actB[t], linW, linb, out + ofs, Ns[t]);
    ofs += (long)Ns[t] * NOUT;
  }
}
